// DifferentialAttention_67645734912848
// MI455X (gfx1250) — matrix operand order repaired and checked by reading, not compiled, not run on hardware
//
#include <hip/hip_runtime.h>
#include <cstdint>
#include <cstddef>

// ---------------------------------------------------------------------------
// CDNA5 (gfx1250) fused differential attention, bf16 WMMA everywhere.
//   Pass 0: fp32 -> bf16 conversion of activations + weights (one pass).
//   Pass 1: QKV projection GEMMs (bf16 in/out), async-LDS double buffer.
//   Pass 2: fused flash-style differential attention; K tiles via the Tensor
//           Data Mover (TENSOR_LOAD_TO_LDS), V transposed in registers->LDS,
//           V fragments shared by both differential streams.
//   Pass 3: output projection GEMM (bf16 in, fp32 out).
// Workspace: 32M bf16 elements = 64 MB in d_ws.
// ---------------------------------------------------------------------------

typedef __attribute__((ext_vector_type(16))) __bf16 v16bf;
typedef __attribute__((ext_vector_type(8)))  float  v8f;
typedef __attribute__((ext_vector_type(4)))  int    v4i;

union Frag16 { v16bf v; unsigned u[8]; };
union Frag8f { v8f   v; float    f[8]; };

__device__ __forceinline__ unsigned short f2bf(float x) {
    unsigned u = __builtin_bit_cast(unsigned, x);
    u += 0x7FFFu + ((u >> 16) & 1u);          // round-to-nearest-even
    return (unsigned short)(u >> 16);
}
__device__ __forceinline__ unsigned pk2(float a, float b) {
    return (unsigned)f2bf(a) | ((unsigned)f2bf(b) << 16);
}

// ---- gfx1250 async global->LDS copy (ASYNCcnt) with safe fallback ---------
#if defined(__HIP_DEVICE_COMPILE__) && \
    __has_builtin(__builtin_amdgcn_global_load_async_to_lds_b128)
#define ASYNC_LDS 1
#endif

__device__ __forceinline__ void cp16_g2l(const void* g, void* l) {
#ifdef ASYNC_LDS
    __builtin_amdgcn_global_load_async_to_lds_b128(
        (__attribute__((address_space(1))) v4i*)g,
        (__attribute__((address_space(3))) v4i*)l, 0, 0);
#else
    *(uint4*)l = *(const uint4*)g;
#endif
}
template <int N>
__device__ __forceinline__ void wait_async() {
#ifdef ASYNC_LDS
#if __has_builtin(__builtin_amdgcn_s_wait_asynccnt)
    __builtin_amdgcn_s_wait_asynccnt(N);
#else
    asm volatile("s_wait_asynccnt %0" :: "i"(N) : "memory");
#endif
#endif
}

// ---- gfx1250 Tensor Data Mover (TENSORcnt) with async fallback ------------
#if defined(__HIP_DEVICE_COMPILE__) && \
    __has_builtin(__builtin_amdgcn_tensor_load_to_lds)
#define HAVE_TDM 1
#endif

#ifdef HAVE_TDM
typedef __attribute__((ext_vector_type(4))) unsigned tdm_u4;
typedef __attribute__((ext_vector_type(8))) int      tdm_i8;
typedef __attribute__((ext_vector_type(4))) int      tdm_i4;

// 2D bf16 tile load: tile_x elems/row, tile_y rows, row stride in elements.
// D# layout per CDNA5 ISA.
__device__ __forceinline__ void tdm_load_2d_bf16(
    const void* gsrc, unsigned lds_addr,
    unsigned tile_x, unsigned tile_y, unsigned row_stride)
{
    unsigned long long ga = (unsigned long long)(size_t)gsrc;
    union { tdm_u4 v; unsigned w[4]; } g0;
    g0.w[0] = 1u;                                            // count=1 (user)
    g0.w[1] = lds_addr;                                      // LDS byte addr
    g0.w[2] = (unsigned)(ga & 0xFFFFFFFFu);                  // global[31:0]
    g0.w[3] = (unsigned)((ga >> 32) & 0x1FFFFFFu) | (2u << 30); // [56:32]|type=2
    union { tdm_i8 v; unsigned w[8]; } g1;
    g1.w[0] = (1u << 16);                                    // data_size=2B
    g1.w[1] = (tile_x << 16);                                // tensor_dim0 lo16
    g1.w[2] = (tile_x >> 16) | (tile_y << 16);               // d0 hi | d1 lo
    g1.w[3] = (tile_y >> 16) | (tile_x << 16);               // d1 hi | tile_dim0
    g1.w[4] = tile_y;                                        // tile_dim1 (d2=0)
    g1.w[5] = row_stride;                                    // dim0_stride lo32
    g1.w[6] = 0u;
    g1.w[7] = 0u;
    tdm_i4 z4 = {0, 0, 0, 0};
    tdm_i8 z8 = {0, 0, 0, 0, 0, 0, 0, 0};
    // 6-arg form (clang-23 / therock HIP headers): groups 0..3 + extra + cpol
    __builtin_amdgcn_tensor_load_to_lds(g0.v, g1.v, z4, z4, z8, 0);
}
#endif

template <int N>
__device__ __forceinline__ void wait_tensor() {
#ifdef HAVE_TDM
#if __has_builtin(__builtin_amdgcn_s_wait_tensorcnt)
    __builtin_amdgcn_s_wait_tensorcnt((short)N);
#else
    asm volatile("s_wait_tensorcnt %0" :: "i"(N) : "memory");
#endif
#endif
}

// ---------------------------------------------------------------------------
// Pass 0: elementwise fp32 -> bf16 (8 elements / thread).
// ---------------------------------------------------------------------------
__global__ __launch_bounds__(256) void cvt_f32_to_bf16(
    const float* __restrict__ src, unsigned short* __restrict__ dst, int n8)
{
    int i = blockIdx.x * 256 + threadIdx.x;
    if (i >= n8) return;
    const float4* s = (const float4*)src + (size_t)i * 2;
    float4 a = s[0], b = s[1];
    unsigned* d = (unsigned*)(dst + (size_t)i * 8);
    d[0] = pk2(a.x, a.y); d[1] = pk2(a.z, a.w);
    d[2] = pk2(b.x, b.y); d[3] = pk2(b.z, b.w);
}

// ---------------------------------------------------------------------------
// C[M,N] = A[M,K] @ W[N,K]^T + bias.   BM=128, BN=64, BK=32, bf16 operands.
// 256 threads = 8 wave32s (4Mx2N); 2x2 WMMA 16x16 tiles per wave.
// LDS double-buffered; tiles streamed with GLOBAL_LOAD_ASYNC_TO_LDS_B128.
// ---------------------------------------------------------------------------
template <bool OUT_BF16>
__global__ __launch_bounds__(256) void gemm_bf16_wmma(
    const unsigned short* __restrict__ A, const unsigned short* __restrict__ W,
    const float* __restrict__ bias, void* __restrict__ Cv,
    int M, int N, int K)
{
    __shared__ unsigned short sA[2][128 * 32];
    __shared__ unsigned short sW[2][64 * 32];

    const int tid  = threadIdx.x;
    const int lane = tid & 31;
    const int wv   = tid >> 5;
    const int wm   = wv & 3;
    const int wn   = wv >> 2;
    const int m0   = blockIdx.y * 128;
    const int n0   = blockIdx.x * 64;

    const int ml  = lane & 15;
    const int kbA = (lane < 16) ? 0 : 16;   // A-operand K base (contiguous k = 16h+s)
    const int kbB = (lane < 16) ? 0 : 16;   // B-operand K base (contiguous k = 16h+s)

    Frag8f acc[2][2];
    for (int mt = 0; mt < 2; ++mt)
        for (int nt = 0; nt < 2; ++nt)
            for (int r = 0; r < 8; ++r) acc[mt][nt].f[r] = 0.0f;

    auto stage = [&](int b, int k0) {
        for (int j = 0; j < 2; ++j) {                 // A: 512 chunks, 2/thread
            int c = tid + j * 256;
            int r = c >> 2, o8 = (c & 3) << 3;
            cp16_g2l(A + (size_t)(m0 + r) * K + k0 + o8, &sA[b][r * 32 + o8]);
        }
        {                                             // W: 256 chunks, 1/thread
            int r = tid >> 2, o8 = (tid & 3) << 3;
            cp16_g2l(W + (size_t)(n0 + r) * K + k0 + o8, &sW[b][r * 32 + o8]);
        }
    };

    const int nk = K / 32;
    stage(0, 0);
    for (int i = 0; i < nk; ++i) {
        const int buf = i & 1;
        if (i + 1 < nk) { stage(buf ^ 1, (i + 1) * 32); wait_async<3>(); }
        else             wait_async<0>();
        __syncthreads();

        Frag16 af[2], bn[2];
        for (int mt = 0; mt < 2; ++mt) {
            int row = wm * 32 + mt * 16 + ml;
            for (int v = 0; v < 8; ++v) {
                int k = kbA + 2 * v;
                af[mt].u[v] = *(const unsigned*)&sA[buf][row * 32 + k];
            }
        }
        for (int nt = 0; nt < 2; ++nt) {
            int col = wn * 32 + nt * 16 + ml;
            for (int v = 0; v < 8; ++v)
                bn[nt].u[v] = *(const unsigned*)&sW[buf][col * 32 + kbB + 2 * v];
        }
        for (int mt = 0; mt < 2; ++mt)
            for (int nt = 0; nt < 2; ++nt)
                acc[mt][nt].v = __builtin_amdgcn_wmma_f32_16x16x32_bf16(
                    false, af[mt].v, false, bn[nt].v,
                    (short)0, acc[mt][nt].v, false, false);
        __syncthreads();
    }

    for (int mt = 0; mt < 2; ++mt)
        for (int nt = 0; nt < 2; ++nt) {
            int col = n0 + wn * 32 + nt * 16 + ml;
            float bvl = bias[col];
            for (int r = 0; r < 8; ++r) {
                int row = m0 + wm * 32 + mt * 16 + ((lane < 16) ? r : r + 8);
                float val = acc[mt][nt].f[r] + bvl;
                if constexpr (OUT_BF16)
                    ((unsigned short*)Cv)[(size_t)row * N + col] = f2bf(val);
                else
                    ((float*)Cv)[(size_t)row * N + col] = val;
            }
        }
}

// ---------------------------------------------------------------------------
// Pass 2: fused differential attention (flash-style, two softmax streams).
// 128 threads = 4 waves; wave w owns 16 query rows of a 64-row q-tile.
// K tiles: one TDM descriptor per tile, issued by wave 0, gated with
// s_wait_tensorcnt (double buffered, <=1 with next tile in flight).
// V tiles: global loads before compute, scatter-transpose [d][s] after.
// ---------------------------------------------------------------------------
#define TSEQ 2048
#define DEMB 1024
#define SCALE_QK 0.17677669529663687f   // 1/sqrt(32)
#define LAMBDA_INIT_V 0.2f              // 0.8 - 0.6*exp(0)

__global__ __launch_bounds__(128) void diff_attn_fused(
    const unsigned short* __restrict__ Q, const unsigned short* __restrict__ Kg,
    const unsigned short* __restrict__ Vg,
    const float* __restrict__ lq1, const float* __restrict__ lk1,
    const float* __restrict__ lq2, const float* __restrict__ lk2,
    unsigned short* __restrict__ Oa)
{
    __shared__ unsigned short sK [2][64 * 64];     // [s][dim] bf16
    __shared__ unsigned short sVT[2][64 * 64];     // [dim][s] bf16 (transposed)
    __shared__ unsigned short sP [4][2][16 * 64];  // per-wave, per-stream P

    const int tid  = threadIdx.x;
    const int lane = tid & 31;
    const int wv   = tid >> 5;
    const int qt   = blockIdx.x;
    const int h    = blockIdx.y;
    const int bb   = blockIdx.z;

    const int ml  = lane & 15;
    const int kbA = (lane < 16) ? 0 : 16;
    const int kbB = (lane < 16) ? 0 : 16;

    const size_t qrow0 = (size_t)bb * TSEQ + (size_t)qt * 64;
    const unsigned short* Qb    = Q  + qrow0 * DEMB + h * 64;
    const unsigned short* Kbase = Kg + ((size_t)bb * TSEQ) * DEMB + h * 64;
    const unsigned short* Vbase = Vg + ((size_t)bb * TSEQ) * DEMB + h * 64;

    // Q fragments (A-operand): q1 = dims 0..31, q2 = dims 32..63
    Frag16 qa[2];
    for (int v = 0; v < 8; ++v) {
        int k = kbA + 2 * v;
        const unsigned short* p = Qb + (size_t)(wv * 16 + ml) * DEMB + k;
        qa[0].u[v] = *(const unsigned*)p;
        qa[1].u[v] = *(const unsigned*)(p + 32);
    }

    float mrow[2][8], lrow[2][8];
    Frag8f oacc[2][4];
    for (int p = 0; p < 2; ++p)
        for (int r = 0; r < 8; ++r) { mrow[p][r] = -1e30f; lrow[p][r] = 0.0f; }
    for (int p = 0; p < 2; ++p)
        for (int j = 0; j < 4; ++j)
            for (int r = 0; r < 8; ++r) oacc[p][j].f[r] = 0.0f;

    // ---- staging helpers ----
    uint4 vreg[4];
    auto issueK = [&](int b, int s0) {
#ifdef HAVE_TDM
        if (wv == 0) {
            unsigned lds_addr = (unsigned)(size_t)
                (__attribute__((address_space(3))) unsigned short*)&sK[b][0];
            tdm_load_2d_bf16(Kbase + (size_t)s0 * DEMB, lds_addr,
                             64u, 64u, (unsigned)DEMB);
        }
#else
        const unsigned short* Kb = Kbase + (size_t)s0 * DEMB;
        for (int j = 0; j < 4; ++j) {
            int c = tid + j * 128;
            int r = c >> 3, o8 = (c & 7) << 3;
            cp16_g2l(Kb + (size_t)r * DEMB + o8, &sK[b][r * 64 + o8]);
        }
#endif
    };
    auto waitK = [&](bool prefetched) {
#ifdef HAVE_TDM
        if (wv == 0) { if (prefetched) wait_tensor<1>(); else wait_tensor<0>(); }
#else
        if (prefetched) wait_async<4>(); else wait_async<0>();
#endif
    };
    auto loadV = [&](int s0) {
        const unsigned short* Vb = Vbase + (size_t)s0 * DEMB;
        for (int j = 0; j < 4; ++j) {
            int c = tid + j * 128;
            int r = c >> 3, o8 = (c & 7) << 3;
            vreg[j] = *(const uint4*)(Vb + (size_t)r * DEMB + o8);
        }
    };
    auto storeV = [&](int b) {          // scatter-transpose into [d][s]
        for (int j = 0; j < 4; ++j) {
            int c = tid + j * 128;
            int r = c >> 3, o8 = (c & 7) << 3;
            union { uint4 q; unsigned short hh[8]; } u; u.q = vreg[j];
            for (int e = 0; e < 8; ++e)
                sVT[b][(o8 + e) * 64 + r] = u.hh[e];
        }
    };

    issueK(0, 0);
    loadV(0);
    storeV(0);

    constexpr int NT = TSEQ / 64;
    for (int it = 0; it < NT; ++it) {
        const int  buf = it & 1;
        const bool pre = (it + 1 < NT);
        if (pre) { issueK(buf ^ 1, (it + 1) * 64); loadV((it + 1) * 64); }
        waitK(pre);                     // tile `it`'s K-DMA retired
        __syncthreads();                // staging of tile `it` visible to all

        // ---- logits + online softmax for both streams; build P frags ----
        Frag16 pa[2][2];
        for (int part = 0; part < 2; ++part) {
            const int kofs = part * 32;
            Frag16 bfr[4];              // load all 4 B-fragments up front
            for (int j = 0; j < 4; ++j)
                for (int v = 0; v < 8; ++v)
                    bfr[j].u[v] = *(const unsigned*)
                        &sK[buf][(j * 16 + ml) * 64 + kofs + kbB + 2 * v];
            Frag8f sacc[4];
            for (int j = 0; j < 4; ++j) {
                Frag8f z;
                for (int r = 0; r < 8; ++r) z.f[r] = 0.0f;
                sacc[j].v = __builtin_amdgcn_wmma_f32_16x16x32_bf16(
                    false, qa[part].v, false, bfr[j].v,
                    (short)0, z.v, false, false);
            }
            for (int j = 0; j < 4; ++j)
                for (int r = 0; r < 8; ++r) sacc[j].f[r] *= SCALE_QK;

            float mnew[8];
            for (int r = 0; r < 8; ++r) {
                float t = fmaxf(fmaxf(sacc[0].f[r], sacc[1].f[r]),
                                fmaxf(sacc[2].f[r], sacc[3].f[r]));
                for (int mk = 1; mk <= 8; mk <<= 1)
                    t = fmaxf(t, __shfl_xor(t, mk, 32));   // half-wave rows
                float mn = fmaxf(mrow[part][r], t);
                mnew[r] = mn;
                float fs = __expf(mrow[part][r] - mn);
                mrow[part][r] = mn;
                lrow[part][r] *= fs;
                for (int j = 0; j < 4; ++j) oacc[part][j].f[r] *= fs;
            }
            for (int r = 0; r < 8; ++r) {
                float rs = 0.0f;
                for (int j = 0; j < 4; ++j) {
                    float pv = __expf(sacc[j].f[r] - mnew[r]);
                    sacc[j].f[r] = pv;
                    rs += pv;
                }
                for (int mk = 1; mk <= 8; mk <<= 1)
                    rs += __shfl_xor(rs, mk, 32);
                lrow[part][r] += rs;
            }

            // C-layout -> A-layout transpose of P via per-wave LDS
            for (int j = 0; j < 4; ++j)
                for (int r = 0; r < 8; ++r) {
                    int row = (lane < 16) ? r : r + 8;
                    sP[wv][part][row * 64 + j * 16 + ml] = f2bf(sacc[j].f[r]);
                }
            for (int sf = 0; sf < 2; ++sf)
                for (int v = 0; v < 8; ++v) {
                    int s = sf * 32 + kbA + 2 * v;
                    pa[part][sf].u[v] =
                        *(const unsigned*)&sP[wv][part][ml * 64 + s];
                }
        }

        // ---- O1/O2 += P1/P2 @ V : each V fragment feeds two WMMAs ----
        for (int jd = 0; jd < 4; ++jd) {
            const int d = jd * 16 + ml;
            for (int sf = 0; sf < 2; ++sf) {
                Frag16 vbf;   // contiguous in [d][s] layout -> 2x ds_load_b128
                for (int v = 0; v < 8; ++v)
                    vbf.u[v] = *(const unsigned*)
                        &sVT[buf][d * 64 + sf * 32 + kbB + 2 * v];
                oacc[0][jd].v = __builtin_amdgcn_wmma_f32_16x16x32_bf16(
                    false, pa[0][sf].v, false, vbf.v,
                    (short)0, oacc[0][jd].v, false, false);
                oacc[1][jd].v = __builtin_amdgcn_wmma_f32_16x16x32_bf16(
                    false, pa[1][sf].v, false, vbf.v,
                    (short)0, oacc[1][jd].v, false, false);
            }
        }

        if (pre) storeV(buf ^ 1);       // overlapped V transpose-store
        __syncthreads();                // all reads of `buf` done before reuse
    }

    // ---- lambda + combined epilogue (bf16 out for the out-proj GEMM) ----
    float d1 = 0.0f, d2 = 0.0f;
    for (int i = 0; i < 32; ++i) { d1 += lq1[i] * lk1[i]; d2 += lq2[i] * lk2[i]; }
    const float lam = __expf(d1) - __expf(d2) + LAMBDA_INIT_V;
    const float osc = 1.0f - LAMBDA_INIT_V;

    unsigned short* Ob = Oa + qrow0 * DEMB + h * 64;
    for (int jd = 0; jd < 4; ++jd)
        for (int r = 0; r < 8; ++r) {
            int row = wv * 16 + ((lane < 16) ? r : r + 8);
            float val = (oacc[0][jd].f[r] / lrow[0][r] -
                         lam * oacc[1][jd].f[r] / lrow[1][r]) * osc;
            Ob[(size_t)row * DEMB + jd * 16 + ml] = f2bf(val);
        }
}

// ---------------------------------------------------------------------------
extern "C" void kernel_launch(void* const* d_in, const int* in_sizes, int n_in,
                              void* d_out, int out_size, void* d_ws, size_t ws_size,
                              hipStream_t stream) {
    (void)in_sizes; (void)n_in; (void)out_size; (void)ws_size;

    const float* query = (const float*)d_in[0];
    const float* key   = (const float*)d_in[1];
    const float* value = (const float*)d_in[2];
    const float* ipw   = (const float*)d_in[3];   // (3D, D)
    const float* ipb   = (const float*)d_in[4];
    const float* opw   = (const float*)d_in[5];   // (D, D)
    const float* opb   = (const float*)d_in[6];
    const float* lq1   = (const float*)d_in[7];
    const float* lk1   = (const float*)d_in[8];
    const float* lq2   = (const float*)d_in[9];
    const float* lk2   = (const float*)d_in[10];

    const int M = 2 * 2048, N = 1024, K = 1024;
    const size_t mat  = (size_t)M * N;       // 4M elements
    const size_t wmat = (size_t)N * K;       // 1M elements

    // bf16 workspace layout: 7*mat + 4*wmat = 32M bf16 = 64 MB
    unsigned short* bq  = (unsigned short*)d_ws;  // converted inputs
    unsigned short* bk  = bq  + mat;
    unsigned short* bv  = bk  + mat;
    unsigned short* biw = bv  + mat;              // in_proj_weight bf16 (3*wmat)
    unsigned short* bow = biw + 3 * wmat;         // out_proj_weight bf16
    unsigned short* qp  = bow + wmat;             // projected Q/K/V (bf16)
    unsigned short* kp  = qp  + mat;
    unsigned short* vp  = kp  + mat;
    unsigned short* ap  = vp  + mat;              // attention output (bf16)

    auto cvt = [&](const float* s, unsigned short* d, size_t n) {
        int n8 = (int)(n / 8);
        cvt_f32_to_bf16<<<(n8 + 255) / 256, 256, 0, stream>>>(s, d, n8);
    };
    cvt(query, bq, mat);
    cvt(key,   bk, mat);
    cvt(value, bv, mat);
    cvt(ipw,   biw, 3 * wmat);
    cvt(opw,   bow, wmat);

    dim3 gg(N / 64, M / 128);                     // (16, 32)
    gemm_bf16_wmma<true ><<<gg, 256, 0, stream>>>(bq, biw,            ipb,         qp, M, N, K);
    gemm_bf16_wmma<true ><<<gg, 256, 0, stream>>>(bk, biw +     wmat, ipb + N,     kp, M, N, K);
    gemm_bf16_wmma<true ><<<gg, 256, 0, stream>>>(bv, biw + 2 * wmat, ipb + 2 * N, vp, M, N, K);

    diff_attn_fused<<<dim3(TSEQ / 64, 16, 2), 128, 0, stream>>>(
        qp, kp, vp, lq1, lk1, lq2, lk2, ap);

    gemm_bf16_wmma<false><<<gg, 256, 0, stream>>>(ap, bow, opb, d_out, M, N, K);
}
